// KOALAModule_82033875354105
// MI455X (gfx1250) — matrix operand order repaired and checked by reading, not compiled, not run on hardware
//
#include <hip/hip_runtime.h>
#include <hip/hip_bf16.h>

typedef __attribute__((ext_vector_type(16))) _Float16 v16h;
typedef __attribute__((ext_vector_type(8)))  _Float16 v8h;
typedef __attribute__((ext_vector_type(8)))  float    v8f;

// ---------------------------------------------------------------------------
// Implicit-GEMM conv (KSxKS, pad=PAD) via v_wmma_f32_16x16x32_f16.
//   out[b,o,p] = bias[o] + sum_{c,tap} w[o,c,tap] * relu?(in[b,c,p+tap])
// Block: 256 thr = 8 waves (4 M x 2 N). Block tile: 64 Mout x 64 pixels.
// ALL weights staged to LDS once (64 x NTAP*64 halves; 72 KB for 3x3).
// Input staged ONCE as a KS-row x (64+KS-1)-col x 64-ch halo in transposed
// [pixel][k] layout (28.5 KB for 3x3): a tap shift is just a row-index
// change, so K-contiguity (2x ds_load_b128 per fragment) is preserved and
// the whole kernel needs exactly ONE barrier. 36 WMMAs issue back-to-back.
// ---------------------------------------------------------------------------
template<int KS, int PAD>
__global__ __launch_bounds__(256)
void conv_wmma(const float* __restrict__ in, const float* __restrict__ w,
               const float* __restrict__ bias, float* __restrict__ out,
               int Cout, int relu_in)
{
    constexpr int Cin = 64, Hh = 128, Ww = 128, HW = Hh * Ww;
    constexpr int NTAP = KS * KS;
    constexpr int AK   = NTAP * 64;            // A row length (halves)
    constexpr int COLS = 64 + (KS - 1);        // halo width (pixels)
    constexpr int BK   = 72;                   // padded halo row stride (halves)

    __shared__ __align__(16) _Float16 a_lds[64 * AK];        // [m][tap*64+c]
    __shared__ __align__(16) _Float16 b_lds[KS * COLS * BK]; // [row][col][k]

    const int tid    = threadIdx.x;
    const int lane   = tid & 31;
    const int wave   = tid >> 5;
    const int wave_m = wave & 3;               // 16 output rows
    const int wave_n = wave >> 2;              // 32-pixel half of the N tile
    const int batch  = blockIdx.y;
    const int pix0   = blockIdx.x * 64;        // 64-pixel strip in one image row
    const int py     = pix0 >> 7;              // image row of the strip
    const int x0     = pix0 & 127;             // starting column of the strip

    const int row  = wave_m * 16 + (lane & 15);
    const int l8   = (lane & 16);              // 0 or 16 (A K-offset)
    const int o16  = (lane & 16);              // 0 or 16 (B K-offset)
    const int col0 = wave_n * 32 + (lane & 15);

    // ---- stage ALL weights once: a_lds[m][tap*64 + c] ----
    for (int i = tid; i < 64 * AK; i += 256) {
        int m  = i / AK;
        int kt = i - m * AK;
        int c  = kt & 63, tap = kt >> 6;
        float v = (m < Cout) ? w[(m * Cin + c) * NTAP + tap] : 0.0f;
        a_lds[i] = (_Float16)v;
    }

    // ---- stage input halo once: transposed [row][col][k] ----
    for (int i = tid; i < KS * COLS * 64; i += 256) {
        int cc = i % COLS;
        int t2 = i / COLS;
        int k  = t2 & 63;
        int rr = t2 >> 6;
        int yy = py + rr - PAD;
        int xx = x0 + cc - PAD;
        float v = 0.0f;
        if (yy >= 0 && yy < Hh && xx >= 0 && xx < Ww) {
            v = in[(batch * Cin + k) * HW + yy * Ww + xx];
            if (relu_in) v = fmaxf(v, 0.0f);
        }
        b_lds[(rr * COLS + cc) * BK + k] = (_Float16)v;
    }
    __syncthreads();      // the only barrier in the kernel

    v8f acc0 = {}, acc1 = {};

    for (int tap = 0; tap < NTAP; ++tap) {
        const int ry = tap / KS, rx = tap % KS;
        const _Float16* arow = a_lds + row * AK + tap * 64;
        const _Float16* brow = b_lds + (ry * COLS + rx) * BK;   // + col*BK below

        #pragma unroll
        for (int kk = 0; kk < 64; kk += 32) {
            v8h alo = *(const v8h*)&arow[kk + l8];
            v8h ahi = *(const v8h*)&arow[kk + l8 + 8];
            v16h af = __builtin_shufflevector(alo, ahi,
                0, 1, 2, 3, 4, 5, 6, 7, 8, 9, 10, 11, 12, 13, 14, 15);

            v8h blo0 = *(const v8h*)&brow[col0 * BK + kk + o16];
            v8h bhi0 = *(const v8h*)&brow[col0 * BK + kk + o16 + 8];
            v16h bf0 = __builtin_shufflevector(blo0, bhi0,
                0, 1, 2, 3, 4, 5, 6, 7, 8, 9, 10, 11, 12, 13, 14, 15);
            acc0 = __builtin_amdgcn_wmma_f32_16x16x32_f16(
                false, af, false, bf0, (short)0, acc0, false, false);

            v8h blo1 = *(const v8h*)&brow[(col0 + 16) * BK + kk + o16];
            v8h bhi1 = *(const v8h*)&brow[(col0 + 16) * BK + kk + o16 + 8];
            v16h bf1 = __builtin_shufflevector(blo1, bhi1,
                0, 1, 2, 3, 4, 5, 6, 7, 8, 9, 10, 11, 12, 13, 14, 15);
            acc1 = __builtin_amdgcn_wmma_f32_16x16x32_f16(
                false, af, false, bf1, (short)0, acc1, false, false);
        }
    }

    // ---- store D: VGPR r -> M = 16*wave_m + r (+8 for lanes 16..31) ----
    #pragma unroll
    for (int r = 0; r < 8; ++r) {
        int o = wave_m * 16 + r + ((lane & 16) ? 8 : 0);
        if (o < Cout) {
            size_t base = ((size_t)batch * Cout + o) * HW + pix0;
            out[base + wave_n * 32 + (lane & 15)]      = acc0[r] + bias[o];
            out[base + wave_n * 32 + 16 + (lane & 15)] = acc1[r] + bias[o];
        }
    }
}

// ---------------------------------------------------------------------------
// feat = fp * mp (elementwise, float4 vectorized)
// ---------------------------------------------------------------------------
__global__ void feat_mul(const float4* __restrict__ a, const float4* __restrict__ b,
                         float4* __restrict__ o, int n4)
{
    int i = blockIdx.x * 256 + threadIdx.x;
    if (i < n4) {
        float4 va = a[i], vb = b[i];
        o[i] = make_float4(va.x * vb.x, va.y * vb.y, va.z * vb.z, va.w * vb.w);
    }
}

// ---------------------------------------------------------------------------
// Final: normalize 49 taps per pixel (registers), 7x7 dynamic local conv with
// LDS-tiled feat halo (22x22 per channel) + prefetch of next channel, residual.
// Block = 16x16 pixel tile, grid = (8,8,B).
// ---------------------------------------------------------------------------
__global__ __launch_bounds__(256)
void koala_final(const float* __restrict__ x, const float* __restrict__ feat,
                 const float* __restrict__ lp, float* __restrict__ out)
{
    constexpr int C = 64, Hh = 128, Ww = 128, HW = Hh * Ww;
    __shared__ float f_s[22 * 22];

    const int tid = threadIdx.x;
    const int tx  = tid & 15, ty = tid >> 4;
    const int bx  = blockIdx.x * 16, by = blockIdx.y * 16;
    const int batch = blockIdx.z;
    const int px = bx + tx, py = by + ty;
    const int p  = py * Ww + px;

    // kernel_normalize: zero-mean over tap dim + 1/49, kept in 49 registers
    float taps[49];
    float s = 0.0f;
    #pragma unroll
    for (int t = 0; t < 49; ++t) {
        float v = lp[((size_t)batch * 49 + t) * HW + p];
        taps[t] = v;
        s += v;
    }
    const float adj = (1.0f / 49.0f) - s * (1.0f / 49.0f);
    #pragma unroll
    for (int t = 0; t < 49; ++t) taps[t] += adj;

    for (int c = 0; c < C; ++c) {
        const float* fch = feat + ((size_t)batch * C + c) * HW;
        // stage 22x22 halo tile; prefetch next channel's tile into cache
        for (int i = tid; i < 22 * 22; i += 256) {
            int ly = i / 22, lx = i % 22;
            int gy = by + ly - 3, gx = bx + lx - 3;
            bool ok = (gy >= 0 && gy < Hh && gx >= 0 && gx < Ww);
            f_s[i] = ok ? fch[gy * Ww + gx] : 0.0f;
            if (ok && c + 1 < C) __builtin_prefetch(fch + HW + gy * Ww + gx, 0, 0);
        }
        __syncthreads();

        float sum = 0.0f;
        #pragma unroll
        for (int dy = 0; dy < 7; ++dy)
            #pragma unroll
            for (int dx = 0; dx < 7; ++dx)
                sum = fmaf(f_s[(ty + dy) * 22 + (tx + dx)], taps[dy * 7 + dx], sum);

        const size_t oidx = ((size_t)batch * C + c) * HW + p;
        out[oidx] = x[oidx] + sum;
        __syncthreads();
    }
}

// ---------------------------------------------------------------------------
extern "C" void kernel_launch(void* const* d_in, const int* in_sizes, int n_in,
                              void* d_out, int out_size, void* d_ws, size_t ws_size,
                              hipStream_t stream)
{
    (void)in_sizes; (void)n_in; (void)out_size; (void)ws_size;

    const float* x   = (const float*)d_in[0];
    const float* kf  = (const float*)d_in[1];
    const float* fw1 = (const float*)d_in[2];  const float* fb1 = (const float*)d_in[3];
    const float* fw2 = (const float*)d_in[4];  const float* fb2 = (const float*)d_in[5];
    const float* mw1 = (const float*)d_in[6];  const float* mb1 = (const float*)d_in[7];
    const float* mw2 = (const float*)d_in[8];  const float* mb2 = (const float*)d_in[9];
    const float* lw1 = (const float*)d_in[10]; const float* lb1 = (const float*)d_in[11];
    const float* lw2 = (const float*)d_in[12]; const float* lb2 = (const float*)d_in[13];
    float* out = (float*)d_out;

    const size_t NB = 4ull * 64 * 128 * 128;   // one BCHW fp32 tensor
    float* t1 = (float*)d_ws;
    float* t2 = t1 + NB;
    float* t3 = t2 + NB;

    dim3 cgrid(256, 4);            // 16384/64 pixel tiles x B
    dim3 cblk(256);

    // feature path: relu -> conv3 -> relu -> conv3
    conv_wmma<3, 1><<<cgrid, cblk, 0, stream>>>(x,  fw1, fb1, t1, 64, 1);
    conv_wmma<3, 1><<<cgrid, cblk, 0, stream>>>(t1, fw2, fb2, t2, 64, 1);   // fp -> t2
    // multiplicative path: conv3 -> relu -> conv3
    conv_wmma<3, 1><<<cgrid, cblk, 0, stream>>>(kf, mw1, mb1, t1, 64, 0);
    conv_wmma<3, 1><<<cgrid, cblk, 0, stream>>>(t1, mw2, mb2, t3, 64, 1);   // mp -> t3
    // feat = fp * mp -> t1
    int n4 = (int)(NB / 4);
    feat_mul<<<(n4 + 255) / 256, 256, 0, stream>>>((const float4*)t2, (const float4*)t3,
                                                   (float4*)t1, n4);
    // local filter path: conv1 -> relu -> conv1 (49 ch)
    conv_wmma<1, 0><<<cgrid, cblk, 0, stream>>>(kf, lw1, lb1, t2, 64, 0);
    conv_wmma<1, 0><<<cgrid, cblk, 0, stream>>>(t2, lw2, lb2, t3, 49, 1);   // lp -> t3
    // normalize + dynamic local conv + residual
    koala_final<<<dim3(8, 8, 4), 256, 0, stream>>>(x, t1, t3, out);
}
